// LinformerCrossAttention_87591563034718
// MI455X (gfx1250) — matrix operand order repaired and checked by reading, not compiled, not run on hardware
//
#include <hip/hip_runtime.h>

typedef __bf16 bf16;
typedef __attribute__((ext_vector_type(2)))  __bf16 v2bf;
typedef __attribute__((ext_vector_type(16))) __bf16 v16bf;
typedef __attribute__((ext_vector_type(8)))  float  v8f;

#define WMMA_BF16(a, b, c) \
  __builtin_amdgcn_wmma_f32_16x16x32_bf16(false, (a), false, (b), (short)0, (c), false, false)

#define NSPLIT 8   // split-K factor for the 8192-deep reduction
#define KS 40      // padded stride for 32-deep staging tiles (80B rows: 16B-aligned, bank-skewed)

// Pack two floats -> two bf16 -> single 32-bit LDS store (for transposing stagers)
static __device__ inline void stpk(bf16* p, float a, float b) {
  v2bf v; v[0] = (bf16)a; v[1] = (bf16)b;
  *(v2bf*)p = v;
}

// ---- WMMA fragment gathers from LDS (contiguous k = 16*half + e in both A and B, wave32) ----------
// A fragment: 16x32 bf16 tile at (row0, k0) of row-major lds[..][ld]
// lane m = lane&15, K = e + 16*half -> 16 contiguous bf16 -> 2x ds_load_b128
static __device__ inline v16bf ldsA(const bf16* base, int row0, int k0, int ld) {
  int lane = threadIdx.x & 31;
  int m = lane & 15, half = lane >> 4;
  const bf16* p = base + (size_t)(row0 + m) * ld + k0 + (half << 4);
  v16bf a;
#pragma unroll
  for (int e = 0; e < 16; ++e) {
    int k = e;
    a[e] = p[k];
  }
  return a;
}
// B fragment from TRANSPOSED staging (lds layout [N][K], stride ldk):
// lane n = lane&15, K = e + 16*half -> 16 contiguous bf16 -> 2x ds_load_b128
static __device__ inline v16bf ldsBt(const bf16* base, int k0, int col0, int ldk) {
  int lane = threadIdx.x & 31;
  int n = lane & 15, half = lane >> 4;
  const bf16* p = base + (size_t)(col0 + n) * ldk + k0 + (half << 4);
  v16bf b;
#pragma unroll
  for (int e = 0; e < 16; ++e) b[e] = p[e];
  return b;
}

// ---- Kernel 1: GX = [E|F]^T @ x2, split-K partials -------------------------
// GX rows 0..255 = E^T x2, rows 256..511 = F^T x2.  Block: 128x128 C tile,
// 8 waves in 2x4, wave tile 64x32.  grid.x = 4 mtiles * 2 ntiles, grid.y = NSPLIT.
__global__ __launch_bounds__(256) void k_gx_partial(
    const float* __restrict__ x2, const float* __restrict__ E,
    const float* __restrict__ F, float* __restrict__ P) {
  __shared__ bf16 As[128 * KS];   // [m][k] (transposed from G)
  __shared__ bf16 Bs[128 * KS];   // [n][k] (transposed from x2 chunk)
  int t = threadIdx.x, wave = t >> 5;
  int wm = wave >> 2, wn = wave & 3;
  int mt = blockIdx.x >> 1, nt = blockIdx.x & 1;
  int m0 = mt * 128, n0 = nt * 128;
  const float* src = (m0 < 256) ? E : F;
  int colbase = m0 & 255;
  int k0 = blockIdx.y * (8192 / NSPLIT);
  int k2 = t >> 4;                 // k-pair id (0..15)
  int mb = (t & 15) << 3;          // 8-wide column strip
  v8f acc[4][2] = {};
  for (int kc = 0; kc < 8192 / NSPLIT; kc += 32) {
    {
      const float* g0 = src + (size_t)(k0 + kc + 2 * k2) * 256 + colbase + mb;
      const float* g1 = g0 + 256;
#pragma unroll
      for (int i = 0; i < 8; ++i) stpk(&As[(mb + i) * KS + 2 * k2], g0[i], g1[i]);
      const float* x0 = x2 + (size_t)(k0 + kc + 2 * k2) * 256 + n0 + mb;
      const float* x1p = x0 + 256;
#pragma unroll
      for (int i = 0; i < 8; ++i) stpk(&Bs[(mb + i) * KS + 2 * k2], x0[i], x1p[i]);
    }
    __syncthreads();
    v16bf bfr[2];
#pragma unroll
    for (int ni = 0; ni < 2; ++ni) bfr[ni] = ldsBt(Bs, 0, 32 * wn + 16 * ni, KS);
#pragma unroll
    for (int mi = 0; mi < 4; ++mi) {
      v16bf a = ldsA(As, 64 * wm + 16 * mi, 0, KS);
#pragma unroll
      for (int ni = 0; ni < 2; ++ni) acc[mi][ni] = WMMA_BF16(a, bfr[ni], acc[mi][ni]);
    }
    __syncthreads();
  }
  int lane = t & 31, half = lane >> 4, nl = lane & 15;
  float* Pb = P + (size_t)blockIdx.y * 512 * 256;
#pragma unroll
  for (int mi = 0; mi < 4; ++mi)
#pragma unroll
    for (int ni = 0; ni < 2; ++ni)
#pragma unroll
      for (int r = 0; r < 8; ++r) {
        int M = m0 + 64 * wm + 16 * mi + 8 * half + r;
        int N = n0 + 32 * wn + 16 * ni + nl;
        Pb[(size_t)M * 256 + N] = acc[mi][ni][r];
      }
}

// ---- Kernel 2: reduce split-K partials ------------------------------------
__global__ __launch_bounds__(256) void k_gx_reduce(const float* __restrict__ P,
                                                   float* __restrict__ GX) {
  int idx = blockIdx.x * 256 + threadIdx.x;  // 512*256 elements
  float s = 0.f;
#pragma unroll
  for (int sp = 0; sp < NSPLIT; ++sp) s += P[(size_t)sp * 512 * 256 + idx];
  GX[idx] = s;
}

// ---- Kernel 3: k_low = EX@Wk, v_low = FX@Wv (256x256x256 each) -------------
__global__ __launch_bounds__(256) void k_lowrank(
    const float* __restrict__ GX, const float* __restrict__ Wk,
    const float* __restrict__ Wv, float* __restrict__ KL, float* __restrict__ VL) {
  __shared__ bf16 As[128 * KS];   // [m][k]
  __shared__ bf16 Bs[128 * KS];   // [n][k] transposed from W
  int t = threadIdx.x, wave = t >> 5, wm = wave >> 2, wn = wave & 3;
  int which = blockIdx.x >> 2;  // 0 = KL, 1 = VL
  int sub = blockIdx.x & 3;
  int m0 = (sub >> 1) * 128, n0 = (sub & 1) * 128;
  const float* A = GX + (size_t)which * 256 * 256;
  const float* W = which ? Wv : Wk;
  float* Out = which ? VL : KL;
  v8f acc[4][2] = {};
  for (int kk = 0; kk < 256; kk += 32) {
    {
      int m = t >> 1, kb = (t & 1) << 4;
      const float* g = A + (size_t)(m0 + m) * 256 + kk + kb;
#pragma unroll
      for (int i = 0; i < 16; i += 2) stpk(&As[m * KS + kb + i], g[i], g[i + 1]);
      int k2 = t >> 4, nb = (t & 15) << 3;
      const float* w0 = W + (size_t)(kk + 2 * k2) * 256 + n0 + nb;
      const float* w1 = w0 + 256;
#pragma unroll
      for (int i = 0; i < 8; ++i) stpk(&Bs[(nb + i) * KS + 2 * k2], w0[i], w1[i]);
    }
    __syncthreads();
    v16bf bfr[2];
#pragma unroll
    for (int ni = 0; ni < 2; ++ni) bfr[ni] = ldsBt(Bs, 0, 32 * wn + 16 * ni, KS);
#pragma unroll
    for (int mi = 0; mi < 4; ++mi) {
      v16bf a = ldsA(As, 64 * wm + 16 * mi, 0, KS);
#pragma unroll
      for (int ni = 0; ni < 2; ++ni) acc[mi][ni] = WMMA_BF16(a, bfr[ni], acc[mi][ni]);
    }
    __syncthreads();
  }
  int lane = t & 31, half = lane >> 4, nl = lane & 15;
#pragma unroll
  for (int mi = 0; mi < 4; ++mi)
#pragma unroll
    for (int ni = 0; ni < 2; ++ni)
#pragma unroll
      for (int r = 0; r < 8; ++r) {
        int M = m0 + 64 * wm + 16 * mi + 8 * half + r;
        int N = n0 + 32 * wn + 16 * ni + nl;
        Out[(size_t)M * 256 + N] = acc[mi][ni][r];
      }
}

// ---- Kernel 4: fused q-proj -> 8-head attention -> Wo ----------------------
// Block = 64 query rows, 256 threads (8 wave32s). x1 read once, out written
// once; Wq/Wo/KL/VL stay L2-resident across all 512 blocks.
__global__ __launch_bounds__(256) void k_attn(
    const float* __restrict__ x1, const float* __restrict__ Wq,
    const float* __restrict__ Wo, const float* __restrict__ KL,
    const float* __restrict__ VL, float* __restrict__ out) {
  __shared__ bf16 qs[64 * 256];    // q * 1/sqrt(HD), bf16
  __shared__ bf16 ss[64 * 256];    // scores -> unnormalized attn weights
  __shared__ bf16 os[64 * 256];    // concatenated per-head outputs
  __shared__ bf16 Bst[256 * KS];   // B staging: [n(256)][k(40)]; also vh as [32][256]
  __shared__ bf16 Ast[64 * KS];    // A staging for q-proj
  __shared__ float rsum[64];       // 1/rowsum for softmax normalization
  int t = threadIdx.x, wave = t >> 5, lane = t & 31, half = lane >> 4, nl = lane & 15;
  size_t r0 = (size_t)blockIdx.x * 64;
  const float scale = 0.17677669529663687f;  // 1/sqrt(32)

  // Step 1: q = x1_tile @ Wq (scaled).  Waves 1x8 over N; wave tile 64x32.
  {
    v8f acc[4][2] = {};
    for (int kk = 0; kk < 256; kk += 32) {
      {
        int r = t >> 2, kb = (t & 3) << 3;
        const float* g = x1 + (r0 + r) * 256 + kk + kb;
#pragma unroll
        for (int i = 0; i < 8; i += 2) stpk(&Ast[r * KS + kb + i], g[i], g[i + 1]);
        int k2 = t >> 4, nb = (t & 15) << 4;
        const float* w0 = Wq + (size_t)(kk + 2 * k2) * 256 + nb;
        const float* w1 = w0 + 256;
        if (kk < 224) __builtin_prefetch(w0 + 512, 0, 0);
#pragma unroll
        for (int i = 0; i < 16; ++i) stpk(&Bst[(nb + i) * KS + 2 * k2], w0[i], w1[i]);
      }
      __syncthreads();
      v16bf bfr[2];
#pragma unroll
      for (int ni = 0; ni < 2; ++ni) bfr[ni] = ldsBt(Bst, 0, 32 * wave + 16 * ni, KS);
#pragma unroll
      for (int mi = 0; mi < 4; ++mi) {
        v16bf a = ldsA(Ast, 16 * mi, 0, KS);
#pragma unroll
        for (int ni = 0; ni < 2; ++ni) acc[mi][ni] = WMMA_BF16(a, bfr[ni], acc[mi][ni]);
      }
      __syncthreads();
    }
#pragma unroll
    for (int mi = 0; mi < 4; ++mi)
#pragma unroll
      for (int ni = 0; ni < 2; ++ni)
#pragma unroll
        for (int r = 0; r < 8; ++r)
          qs[(16 * mi + 8 * half + r) * 256 + 32 * wave + 16 * ni + nl] =
              (bf16)(acc[mi][ni][r] * scale);
  }
  __syncthreads();

  for (int h = 0; h < 8; ++h) {
    // stage kh^T transposed: Bst[kp][hd] = KL[kp][32h+hd]  (contiguous both ways)
    {
      const float* g = KL + (size_t)t * 256 + h * 32;
#pragma unroll
      for (int j = 0; j < 32; j += 2) stpk(&Bst[t * KS + j], g[j], g[j + 1]);
    }
    __syncthreads();
    // scores = q_h (64x32) @ kh^T (32x256); single K=32 WMMA step
    {
      v8f acc[4][2] = {};
      v16bf bfr[2];
#pragma unroll
      for (int ni = 0; ni < 2; ++ni) bfr[ni] = ldsBt(Bst, 0, 32 * wave + 16 * ni, KS);
#pragma unroll
      for (int mi = 0; mi < 4; ++mi) {
        v16bf a = ldsA(qs, 16 * mi, 32 * h, 256);
#pragma unroll
        for (int ni = 0; ni < 2; ++ni) acc[mi][ni] = WMMA_BF16(a, bfr[ni], acc[mi][ni]);
      }
#pragma unroll
      for (int mi = 0; mi < 4; ++mi)
#pragma unroll
        for (int ni = 0; ni < 2; ++ni)
#pragma unroll
          for (int r = 0; r < 8; ++r)
            ss[(16 * mi + 8 * half + r) * 256 + 32 * wave + 16 * ni + nl] =
                (bf16)acc[mi][ni][r];
    }
    __syncthreads();
    // stage vh transposed: Bst[hd][kp] = VL[kp][32h+hd] (pair-packed along kp)
    if (t < 128) {
      int kp0 = 2 * t;
      const float* g0 = VL + (size_t)kp0 * 256 + h * 32;
      const float* g1 = g0 + 256;
#pragma unroll
      for (int kd = 0; kd < 32; ++kd) stpk(&Bst[kd * 256 + kp0], g0[kd], g1[kd]);
    }
    // softmax: 4 threads per row, wave-local shuffle reduction
    {
      int r = t >> 2, c0 = (t & 3) << 6;
      bf16* row = ss + r * 256 + c0;
      float m = -3.0e38f;
#pragma unroll 16
      for (int c = 0; c < 64; ++c) m = fmaxf(m, (float)row[c]);
      m = fmaxf(m, __shfl_xor(m, 1, 32));
      m = fmaxf(m, __shfl_xor(m, 2, 32));
      float sum = 0.f;
#pragma unroll 16
      for (int c = 0; c < 64; ++c) {
        float e = __expf((float)row[c] - m);
        row[c] = (bf16)e;
        sum += e;
      }
      sum += __shfl_xor(sum, 1, 32);
      sum += __shfl_xor(sum, 2, 32);
      if ((t & 3) == 0) rsum[r] = 1.0f / sum;
    }
    __syncthreads();
    // o_h = attn (64x256) @ vh (256x32); normalization folded into epilogue
    {
      int mi = wave >> 1, ni = wave & 1;
      v8f acc = {};
#pragma unroll
      for (int ks = 0; ks < 8; ++ks) {
        v16bf a = ldsA(ss, 16 * mi, 32 * ks, 256);
        v16bf b = ldsBt(Bst, 32 * ks, 16 * ni, 256);
        acc = WMMA_BF16(a, b, acc);
      }
#pragma unroll
      for (int r = 0; r < 8; ++r) {
        int M = 16 * mi + 8 * half + r;
        os[M * 256 + 32 * h + 16 * ni + nl] = (bf16)(acc[r] * rsum[M]);
      }
    }
    __syncthreads();
  }

  // Step 5: out_tile = os @ Wo (fp32 result to global)
  {
    v8f acc[4][2] = {};
    for (int kk = 0; kk < 256; kk += 32) {
      {
        int k2 = t >> 4, nb = (t & 15) << 4;
        const float* w0 = Wo + (size_t)(kk + 2 * k2) * 256 + nb;
        const float* w1 = w0 + 256;
        if (kk < 224) __builtin_prefetch(w0 + 512, 0, 0);
#pragma unroll
        for (int i = 0; i < 16; ++i) stpk(&Bst[(nb + i) * KS + 2 * k2], w0[i], w1[i]);
      }
      __syncthreads();
      v16bf bfr[2];
#pragma unroll
      for (int ni = 0; ni < 2; ++ni) bfr[ni] = ldsBt(Bst, 0, 32 * wave + 16 * ni, KS);
#pragma unroll
      for (int mi = 0; mi < 4; ++mi) {
        v16bf a = ldsA(os, 16 * mi, kk, 256);
#pragma unroll
        for (int ni = 0; ni < 2; ++ni) acc[mi][ni] = WMMA_BF16(a, bfr[ni], acc[mi][ni]);
      }
      __syncthreads();
    }
#pragma unroll
    for (int mi = 0; mi < 4; ++mi)
#pragma unroll
      for (int ni = 0; ni < 2; ++ni)
#pragma unroll
        for (int r = 0; r < 8; ++r) {
          int M = 16 * mi + 8 * half + r;
          int N = 32 * wave + 16 * ni + nl;
          out[(r0 + M) * 256 + N] = acc[mi][ni][r];
        }
  }
}

// ---- host ------------------------------------------------------------------
extern "C" void kernel_launch(void* const* d_in, const int* in_sizes, int n_in,
                              void* d_out, int out_size, void* d_ws, size_t ws_size,
                              hipStream_t stream) {
  const float* x1 = (const float*)d_in[0];
  const float* x2 = (const float*)d_in[1];
  const float* Wq = (const float*)d_in[2];
  const float* Wk = (const float*)d_in[3];
  const float* Wv = (const float*)d_in[4];
  const float* Wo = (const float*)d_in[5];
  const float* E  = (const float*)d_in[6];
  const float* F  = (const float*)d_in[7];
  float* out = (float*)d_out;
  float* ws = (float*)d_ws;
  float* P  = ws;                              // NSPLIT*512*256 fp32 (4 MB)
  float* GX = P + (size_t)NSPLIT * 512 * 256;  // 512*256
  float* KL = GX + 512 * 256;                  // 256*256
  float* VL = KL + 256 * 256;                  // 256*256

  k_gx_partial<<<dim3(8, NSPLIT), 256, 0, stream>>>(x2, E, F, P);
  k_gx_reduce<<<512, 256, 0, stream>>>(P, GX);
  k_lowrank<<<8, 256, 0, stream>>>(GX, Wk, Wv, KL, VL);
  k_attn<<<32768 / 64, 256, 0, stream>>>(x1, Wq, Wo, KL, VL, out);
}
